// LMMBlock_30253749633469
// MI455X (gfx1250) — compile-verified, matrix operand order checked by reading, not run on hardware
//
#include <hip/hip_runtime.h>
#include <hip/hip_bf16.h>
#include <math.h>

// ---------------------------------------------------------------------------
// LMM block for MI455X (gfx1250, wave32, WMMA).
//   B=4, S=2048, D=1024, F=4096, fp32 in/out.
// GEMMs: bf16 hi/lo split (3x v_wmma_f32_16x16x32_bf16 per K=32) for ~fp32
// accuracy at bf16 matrix-core rate. Recurrence: column-separable Titans scan
// with M/Smom resident in 256KB LDS, and double-buffered
// global_load_async_to_lds_b128 staging of per-step operands (ASYNCcnt).
// ---------------------------------------------------------------------------

typedef __attribute__((ext_vector_type(16))) __bf16 v16bf;
typedef __attribute__((ext_vector_type(8)))  float  v8f;

#define BB 4
#define SS 2048
#define DD 1024
#define FF 4096
#define BS (BB * SS)  // 8192
#define QSTRIDE 3080  // 3072 qkv + 3 gates + pad (16B multiple)

__device__ __forceinline__ unsigned short f2bf(float f) {
  unsigned u = __float_as_uint(f);
  unsigned r = u + 0x7FFFu + ((u >> 16) & 1u); // round-to-nearest-even
  return (unsigned short)(r >> 16);
}
__device__ __forceinline__ float bf2f(unsigned short h) {
  return __uint_as_float(((unsigned)h) << 16);
}

// ---------------------------------------------------------------------------
// Transpose W[K,N] (row-major) -> T[N,K] (row-major), split into bf16 hi/lo.
// ---------------------------------------------------------------------------
__global__ __launch_bounds__(256) void transpose_split(
    const float* __restrict__ W, unsigned short* __restrict__ Th,
    unsigned short* __restrict__ Tl, int K, int N) {
  const int n = blockIdx.x * 256 + threadIdx.x;
  const int k = blockIdx.y;
  const float v = W[(size_t)k * N + n];
  const unsigned short h = f2bf(v);
  Th[(size_t)n * K + k] = h;
  Tl[(size_t)n * K + k] = f2bf(v - bf2f(h));
}

// ---------------------------------------------------------------------------
// RMSNorm one row of D=1024, split result to bf16 hi/lo. 256 thr, 4 elem/thr.
// ---------------------------------------------------------------------------
__global__ __launch_bounds__(256) void rmsnorm_split(
    const float* __restrict__ x, const float* __restrict__ wgt,
    unsigned short* __restrict__ oh, unsigned short* __restrict__ ol) {
  __shared__ float rbuf[256];
  const int row = blockIdx.x;
  const int tid = threadIdx.x;
  const float4 v = ((const float4*)(x + (size_t)row * DD))[tid];
  rbuf[tid] = v.x * v.x + v.y * v.y + v.z * v.z + v.w * v.w;
  __syncthreads();
  for (int o = 128; o; o >>= 1) {
    if (tid < o) rbuf[tid] += rbuf[tid + o];
    __syncthreads();
  }
  const float scale = rsqrtf(rbuf[0] * (1.0f / DD) + 1e-6f);
  const float4 wv = ((const float4*)wgt)[tid];
  float vals[4] = {v.x * scale * wv.x, v.y * scale * wv.y,
                   v.z * scale * wv.z, v.w * scale * wv.w};
  const size_t o0 = (size_t)row * DD + (size_t)tid * 4;
#pragma unroll
  for (int c = 0; c < 4; ++c) {
    unsigned short h = f2bf(vals[c]);
    oh[o0 + c] = h;
    ol[o0 + c] = f2bf(vals[c] - bf2f(h));
  }
}

// ---------------------------------------------------------------------------
// Gate scalars: sigmoid(normed . w_{alpha,eta,theta}); written into the tail
// of the padded qkv row so the recurrence can fetch q|k|v|gates as one
// contiguous async-staged region.
// ---------------------------------------------------------------------------
__global__ __launch_bounds__(256) void gates_kernel(
    const unsigned short* __restrict__ nh, const unsigned short* __restrict__ nl,
    const float* __restrict__ wa, const float* __restrict__ we,
    const float* __restrict__ wt, float* __restrict__ qkv) {
  __shared__ float ra[256], rb[256], rc[256];
  const int row = blockIdx.x;
  const int tid = threadIdx.x;
  const size_t base = (size_t)row * DD;
  float aa = 0.f, ee = 0.f, tt = 0.f;
#pragma unroll
  for (int c = 0; c < 4; ++c) {
    const int i = tid * 4 + c;
    const float v = bf2f(nh[base + i]) + bf2f(nl[base + i]);
    aa += v * wa[i];
    ee += v * we[i];
    tt += v * wt[i];
  }
  ra[tid] = aa; rb[tid] = ee; rc[tid] = tt;
  __syncthreads();
  for (int o = 128; o; o >>= 1) {
    if (tid < o) { ra[tid] += ra[tid + o]; rb[tid] += rb[tid + o]; rc[tid] += rc[tid + o]; }
    __syncthreads();
  }
  if (tid == 0) {
    float* g = qkv + (size_t)row * QSTRIDE + 3072;
    g[0] = 1.f / (1.f + __expf(-ra[0]));  // alpha
    g[1] = 1.f / (1.f + __expf(-rb[0]));  // eta
    g[2] = 1.f / (1.f + __expf(-rc[0]));  // theta
  }
}

// ---------------------------------------------------------------------------
// WMMA GEMM: C[M,N] = (Ah+Al)[M,K] @ (Bh+Bl)^T where B given as [N,K].
// Block = 256 thr = 8 waves; block tile 64(M) x 128(N); wave tile 16x64.
// MODE 0: store fp32 C.  MODE 1: h = silu(G)*acc -> bf16 hi/lo (Hh,Hl).
// MODE 2: C += acc (residual accumulate in place).
// ---------------------------------------------------------------------------
template <int MODE>
__global__ __launch_bounds__(256) void gemm_wmma(
    const unsigned short* __restrict__ Ah, const unsigned short* __restrict__ Al,
    const unsigned short* __restrict__ Bh, const unsigned short* __restrict__ Bl,
    float* __restrict__ C, const float* __restrict__ G,
    unsigned short* __restrict__ Hh, unsigned short* __restrict__ Hl,
    int M, int N, int K, int ldc) {
  const int tid = threadIdx.x;
  const int wave = tid >> 5;
  const int lane = tid & 31;
  const int m0 = blockIdx.y * 64 + (wave & 3) * 16;
  const int n0 = blockIdx.x * 128 + (wave >> 2) * 64;
  const int laneN = lane & 15;
  const int koff = (lane >> 4) << 4;
  const size_t arow = (size_t)(m0 + laneN) * K;

  v8f acc[4] = {v8f{}, v8f{}, v8f{}, v8f{}};

  for (int k = 0; k < K; k += 32) {
    const size_t ao = arow + (size_t)(k + koff);
    __builtin_prefetch(Ah + ao + 128, 0, 1);
    v16bf ah = *(const v16bf*)(Ah + ao);
    v16bf al = *(const v16bf*)(Al + ao);
#pragma unroll
    for (int j = 0; j < 4; ++j) {
      const size_t bo = (size_t)(n0 + j * 16 + laneN) * K + (size_t)(k + koff);
      v16bf bh = *(const v16bf*)(Bh + bo);
      v16bf bl = *(const v16bf*)(Bl + bo);
      acc[j] = __builtin_amdgcn_wmma_f32_16x16x32_bf16(
          false, ah, false, bh, (short)0, acc[j], false, false);
      acc[j] = __builtin_amdgcn_wmma_f32_16x16x32_bf16(
          false, ah, false, bl, (short)0, acc[j], false, false);
      acc[j] = __builtin_amdgcn_wmma_f32_16x16x32_bf16(
          false, al, false, bh, (short)0, acc[j], false, false);
    }
  }

  const int mh = (lane >> 4) * 8;
#pragma unroll
  for (int j = 0; j < 4; ++j) {
    const int ncol = n0 + j * 16 + laneN;
#pragma unroll
    for (int r = 0; r < 8; ++r) {
      const size_t idx = (size_t)(m0 + mh + r) * ldc + ncol;
      const float v = acc[j][r];
      if (MODE == 0) {
        C[idx] = v;
      } else if (MODE == 2) {
        C[idx] += v;
      } else {
        const float g = G[idx];
        const float h = (g / (1.f + __expf(-g))) * v; // silu(gate) * up
        const unsigned short hb = f2bf(h);
        Hh[idx] = hb;
        Hl[idx] = f2bf(h - bf2f(hb));
      }
    }
  }
}

// ---------------------------------------------------------------------------
// Async stage: copy q|k (2048 f32), v-slice (32 f32), gates (3 f32) of row m
// into LDS stage buffer with GLOBAL_LOAD_ASYNC_TO_LDS_B128. Exactly one async
// instruction per issuing wave (waves 0..16), so ASYNCcnt bookkeeping is
// uniform: in-order completion => s_wait_asynccnt 1 retires the older step.
// Stage layout (floats): [0,2048) q|k ; [2048,2080) v-slice ; [2080,2083) gates.
// ---------------------------------------------------------------------------
__device__ __forceinline__ void issue_stage(const float* __restrict__ qkv,
                                            float* stage, size_t m, int e0,
                                            int tid) {
  const float* row = qkv + m * QSTRIDE;
  if (tid < 512) {
    const unsigned lds = (unsigned)(uintptr_t)(stage + tid * 4);
    const unsigned long long ga = (unsigned long long)(uintptr_t)(row + tid * 4);
    asm volatile("global_load_async_to_lds_b128 %0, %1, off"
                 :: "v"(lds), "v"(ga) : "memory");
  } else if (tid < 521) {
    const int j = tid - 512;                       // 0..7 -> v, 8 -> gates
    const int so = (j < 8) ? (2048 + e0 + j * 4) : 3072;
    const int dofs = (j < 8) ? (2048 + j * 4) : 2080;
    const unsigned lds = (unsigned)(uintptr_t)(stage + dofs);
    const unsigned long long ga = (unsigned long long)(uintptr_t)(row + so);
    asm volatile("global_load_async_to_lds_b128 %0, %1, off"
                 :: "v"(lds), "v"(ga) : "memory");
  }
}

// ---------------------------------------------------------------------------
// Titans recurrence, column-separable. Block (b, g) owns batch b, columns
// e0 = 32g .. 32g+31. M[:,e] and Smom[:,e] live in LDS for all 2048 steps.
// 1024 threads = 32 waves; thread (dg, e) owns rows 32dg..32dg+31 of col e0+e.
//   pred = k^T M ; err = pred - v ; S = eta*S - th*k*err ; M = (1-a)*M + S
//   out  = q^T M_new (fused into update pass)
// Per-step operands double-buffered via async LDS staging.
// ---------------------------------------------------------------------------
__global__ __launch_bounds__(1024) void recurrence_kernel(
    const float* __restrict__ qkv,   // [BS, QSTRIDE] : q | k | v | gates
    float* __restrict__ mem_out,     // [BS, 1024]
    float* __restrict__ Mout,        // [B, 1024, 1024]
    float* __restrict__ Sout) {      // [B, 1024, 1024]
  extern __shared__ float smem[];
  float* Ml  = smem;           // 32768
  float* Sl  = Ml + 32768;     // 32768
  float* stg = Sl + 32768;     // 2 * 2088 (16B-aligned stage buffers)
  float* red = stg + 2 * 2088; // 32*33 (padded)
  float* rfin = red + 1056;    // 32

  const int tid = threadIdx.x;
  const int e = tid & 31;   // column within slice
  const int dg = tid >> 5;  // row group (== wave id)
  const int b = blockIdx.x >> 5;
  const int e0 = (blockIdx.x & 31) * 32;

  // kick off step-0 operand staging, overlap with state zero-init
  issue_stage(qkv, stg, (size_t)b * SS, e0, tid);

#pragma unroll 8
  for (int i = 0; i < 32; ++i) {
    Ml[(dg * 32 + i) * 32 + e] = 0.f;
    Sl[(dg * 32 + i) * 32 + e] = 0.f;
  }

  int p = 0;
  for (int t = 0; t < SS; ++t) {
    const size_t m = (size_t)b * SS + t;
    float* stage = stg + p * 2088;
    // prefetch next step into the other buffer, then retire this step's copies
    if (t + 1 < SS) {
      issue_stage(qkv, stg + (p ^ 1) * 2088, m + 1, e0, tid);
      asm volatile("s_wait_asynccnt 0x1" ::: "memory");
    } else {
      asm volatile("s_wait_asynccnt 0x0" ::: "memory");
    }
    __syncthreads();

    const float* sq = stage;         // q_t
    const float* sk = stage + 1024;  // k_t

    // pass 1: pred partial = k^T M over my row group
    float pp = 0.f;
#pragma unroll 8
    for (int i = 0; i < 32; ++i) {
      const int d = dg * 32 + i;
      pp += sk[d] * Ml[d * 32 + e];
    }
    red[dg * 33 + e] = pp;
    __syncthreads();
    if (dg == 0) {
      float tot = 0.f;
#pragma unroll 8
      for (int j = 0; j < 32; ++j) tot += red[j * 33 + e];
      rfin[e] = tot;
    }
    __syncthreads();

    const float a = stage[2080], eta = stage[2081], th = stage[2082];
    const float one_a = 1.f - a;
    const float errv = rfin[e] - stage[2048 + e];

    // pass 2: update state + out partial = q^T M_new
    float op = 0.f;
#pragma unroll 8
    for (int i = 0; i < 32; ++i) {
      const int d = dg * 32 + i;
      const int idx = d * 32 + e;
      float s = Sl[idx];
      float mm = Ml[idx];
      s = eta * s - th * sk[d] * errv;
      mm = one_a * mm + s;
      Sl[idx] = s;
      Ml[idx] = mm;
      op += sq[d] * mm;
    }
    red[dg * 33 + e] = op;
    __syncthreads();
    if (dg == 0) {
      float tot = 0.f;
#pragma unroll 8
      for (int j = 0; j < 32; ++j) tot += red[j * 33 + e];
      mem_out[m * 1024 + e0 + e] = tot;
    }
    __syncthreads();
    p ^= 1;
  }

  // dump final state (coalesced: lane e -> consecutive addresses)
  const size_t base = (size_t)b * (1024 * 1024);
#pragma unroll 4
  for (int i = 0; i < 32; ++i) {
    const int d = dg * 32 + i;
    Mout[base + (size_t)d * 1024 + e0 + e] = Ml[d * 32 + e];
    Sout[base + (size_t)d * 1024 + e0 + e] = Sl[d * 32 + e];
  }
}

__global__ __launch_bounds__(256) void add_kernel(
    const float* __restrict__ a, const float* __restrict__ b,
    float* __restrict__ o) {
  const size_t i = ((size_t)blockIdx.x * blockDim.x + threadIdx.x) * 4;
  const float4 va = *(const float4*)(a + i);
  const float4 vb = *(const float4*)(b + i);
  float4 vo;
  vo.x = va.x + vb.x; vo.y = va.y + vb.y;
  vo.z = va.z + vb.z; vo.w = va.w + vb.w;
  *(float4*)(o + i) = vo;
}

// ---------------------------------------------------------------------------
extern "C" void kernel_launch(void* const* d_in, const int* in_sizes, int n_in,
                              void* d_out, int out_size, void* d_ws,
                              size_t ws_size, hipStream_t stream) {
  (void)in_sizes; (void)n_in; (void)out_size; (void)ws_size;
  const float* x   = (const float*)d_in[0];
  const float* n1w = (const float*)d_in[1];
  const float* n2w = (const float*)d_in[2];
  const float* Wq  = (const float*)d_in[3];
  const float* Wk  = (const float*)d_in[4];
  const float* Wv  = (const float*)d_in[5];
  const float* wa  = (const float*)d_in[6];
  const float* we  = (const float*)d_in[7];
  const float* wt  = (const float*)d_in[8];
  const float* gw  = (const float*)d_in[9];
  const float* uw  = (const float*)d_in[10];
  const float* dw  = (const float*)d_in[11];

  float* xout = (float*)d_out;                 // [8192,1024]
  float* Mout = xout + (size_t)BS * DD;        // [4,1024,1024]
  float* Sout = Mout + (size_t)BB * DD * DD;   // [4,1024,1024]

  // workspace carve (256B aligned)
  char* wbase = (char*)d_ws;
  size_t off = 0;
  auto alloc = [&](size_t bytes) -> void* {
    void* p = wbase + off;
    off = (off + bytes + 255) & ~(size_t)255;
    return p;
  };
  unsigned short* nh  = (unsigned short*)alloc((size_t)BS * DD * 2);
  unsigned short* nl  = (unsigned short*)alloc((size_t)BS * DD * 2);
  unsigned short* wTh = (unsigned short*)alloc((size_t)FF * DD * 2); // max N*K
  unsigned short* wTl = (unsigned short*)alloc((size_t)FF * DD * 2);
  float* big  = (float*)alloc((size_t)BS * FF * 4); // qkv (BS*3080) / gate buf
  float* qkv  = big;
  float* gbuf = big;
  unsigned short* hh = (unsigned short*)alloc((size_t)BS * FF * 2);
  unsigned short* hl = (unsigned short*)alloc((size_t)BS * FF * 2);
  float* memo = (float*)alloc((size_t)BS * DD * 4);

  const dim3 t256(256);

  // 1) fused QKV weight transpose/split: rows [0,1024)=Wq, [1024,2048)=Wk, ...
  transpose_split<<<dim3(DD / 256, DD), t256, 0, stream>>>(Wq, wTh, wTl, DD, DD);
  transpose_split<<<dim3(DD / 256, DD), t256, 0, stream>>>(
      Wk, wTh + (size_t)DD * DD, wTl + (size_t)DD * DD, DD, DD);
  transpose_split<<<dim3(DD / 256, DD), t256, 0, stream>>>(
      Wv, wTh + (size_t)2 * DD * DD, wTl + (size_t)2 * DD * DD, DD, DD);

  // 2) rmsnorm1 -> bf16 hi/lo
  rmsnorm_split<<<dim3(BS), t256, 0, stream>>>(x, n1w, nh, nl);

  // 3) qkv = normed @ [Wq|Wk|Wv]   (M=8192, N=3072, K=1024, ldc=3080)
  gemm_wmma<0><<<dim3(3072 / 128, BS / 64), t256, 0, stream>>>(
      nh, nl, wTh, wTl, qkv, nullptr, nullptr, nullptr, BS, 3072, DD, QSTRIDE);

  // 4) gate scalars -> qkv row tails
  gates_kernel<<<dim3(BS), t256, 0, stream>>>(nh, nl, wa, we, wt, qkv);

  // 5) recurrence: LDS-resident state + async double-buffered staging
  const int SMEM_BYTES = (32768 * 2 + 2 * 2088 + 1056 + 32) * 4;
  hipFuncSetAttribute((const void*)recurrence_kernel,
                      hipFuncAttributeMaxDynamicSharedMemorySize, SMEM_BYTES);
  recurrence_kernel<<<dim3(BB * 32), dim3(1024), SMEM_BYTES, stream>>>(
      qkv, memo, Mout, Sout);

  // 6) x2 = x + mem_out  -> output x slot
  add_kernel<<<dim3((BS * DD) / 1024), t256, 0, stream>>>(x, memo, xout);

  // 7) rmsnorm2 -> bf16 hi/lo
  rmsnorm_split<<<dim3(BS), t256, 0, stream>>>(xout, n2w, nh, nl);

  // 8) gate projection  (M=8192, N=4096, K=1024) -> fp32 gbuf
  transpose_split<<<dim3(FF / 256, DD), t256, 0, stream>>>(gw, wTh, wTl, DD, FF);
  gemm_wmma<0><<<dim3(FF / 128, BS / 64), t256, 0, stream>>>(
      nh, nl, wTh, wTl, gbuf, nullptr, nullptr, nullptr, BS, FF, DD, FF);

  // 9) up projection + fused silu(gate)*up epilogue -> bf16 hi/lo hidden
  transpose_split<<<dim3(FF / 256, DD), t256, 0, stream>>>(uw, wTh, wTl, DD, FF);
  gemm_wmma<1><<<dim3(FF / 128, BS / 64), t256, 0, stream>>>(
      nh, nl, wTh, wTl, nullptr, gbuf, hh, hl, BS, FF, DD, FF);

  // 10) down projection, accumulate into x output (M=8192, N=1024, K=4096)
  transpose_split<<<dim3(DD / 256, FF), t256, 0, stream>>>(dw, wTh, wTl, FF, DD);
  gemm_wmma<2><<<dim3(DD / 128, BS / 64), t256, 0, stream>>>(
      hh, hl, wTh, wTl, xout, nullptr, nullptr, nullptr, BS, DD, FF, DD);
}
